// FeatureQuantizer_25074019074482
// MI455X (gfx1250) — compile-verified, matrix operand order checked by reading, not run on hardware
//
#include <hip/hip_runtime.h>

typedef __attribute__((ext_vector_type(16))) __bf16 v16bf;
typedef __attribute__((ext_vector_type(8)))  float  v8f;
typedef unsigned int v4u __attribute__((ext_vector_type(4)));
typedef int          v8i __attribute__((ext_vector_type(8)));
typedef int          v4i __attribute__((ext_vector_type(4)));

#define EMB_DIM 256
#define NUM_EMB 1024
#define N_ROWS  32768          // 32 * 32 * 32
#define OUT0_ELEMS 8388608     // 32*256*32*32

#if __has_builtin(__builtin_amdgcn_tensor_load_to_lds) && \
    __has_builtin(__builtin_amdgcn_s_wait_tensorcnt)
#define USE_TDM 1
#else
#define USE_TDM 0
#endif

__device__ __forceinline__ unsigned short f2bf(float f) {
    unsigned u = __float_as_uint(f);
    u += 0x7FFFu + ((u >> 16) & 1u);     // round-to-nearest-even
    return (unsigned short)(u >> 16);
}
__device__ __forceinline__ float bf2f(unsigned short h) {
    return __uint_as_float(((unsigned)h) << 16);
}

#if USE_TDM
// 1-D TDM descriptor: copy 8KB (2048 x 4B) from contiguous global to LDS.
// D# group0: [1:0]=count=1, [63:32]=lds_addr, [120:64]=global_addr, [127:126]=type=2
// D# group1: [17:16]=data_size=2 (4B), [79:48]=tensor_dim0=2048,
//            [111:80]=tensor_dim1=1, [127:112]=tile_dim0=2048,
//            [207:160]=tensor_dim0_stride=2048
// This toolchain declares the 6-arg builtin:
//   (uint32x4 g0, int32x8 g1, int32x4 g2, int32x4 g3, int32x8 gx, i32 cpol)
__device__ __forceinline__ void tdm_load_8k(const void* src, unsigned ldsOff) {
    unsigned long long ga = (unsigned long long)(uintptr_t)src;
    v4u g0 = { 1u,
               ldsOff,
               (unsigned)ga,
               ((unsigned)(ga >> 32) & 0x01FFFFFFu) | (2u << 30) };
    v8i g1 = { (int)(2u << 16),        // data_size = 4B
               (int)(2048u << 16),     // tensor_dim0[15:0] in bits 63:48
               (int)(1u << 16),        // tensor_dim0 hi = 0 ; tensor_dim1 = 1
               (int)(2048u << 16),     // tensor_dim1 hi = 0 ; tile_dim0 = 2048
               0,                      // tile_dim1 = 0, tile_dim2 = 0
               2048,                   // tensor_dim0_stride[31:0]
               0, 0 };
    v4i gz4 = { 0, 0, 0, 0 };
    v8i gz8 = { 0, 0, 0, 0, 0, 0, 0, 0 };
    __builtin_amdgcn_tensor_load_to_lds(g0, g1, gz4, gz4, gz8, 0);
}
#endif

// ---------------------------------------------------------------------------
// K1a: split embed into bf16 hi/lo, stored pre-swizzled in WMMA B-fragment
// order so each lane's 16-bf16 fragment is 32 contiguous bytes and each
// 16-code tile is one contiguous 8KB block (TDM-friendly).
// ---------------------------------------------------------------------------
__global__ void k_embed_swizzle(const float* __restrict__ embed,
                                unsigned short* __restrict__ hiOut,
                                unsigned short* __restrict__ loOut) {
    int e = blockIdx.x * 256 + threadIdx.x;        // e = c*1024 + k
    int k = e & 1023, c = e >> 10;
    float f = embed[e];
    unsigned short h = f2bf(f);
    float lo = f - bf2f(h);
    unsigned short l = f2bf(lo);
    int t = k >> 4, j = k & 15, chunk = c >> 5, cc = c & 31;
    int lane = j | (((cc >> 3) & 1) << 4);
    int slot = (cc & 7) | (((cc >> 4) & 1) << 3);
    int o = ((t * 8 + chunk) * 32 + lane) * 16 + slot;
    hiOut[o] = h;
    loOut[o] = l;
}

// K1b: exact f32 ||e_k||^2
__global__ void k_embed_norm(const float* __restrict__ embed,
                             float* __restrict__ enorm) {
    int k = blockIdx.x * 256 + threadIdx.x;        // 0..1023
    float s = 0.f;
    for (int c = 0; c < EMB_DIM; ++c) {
        float v = embed[c * NUM_EMB + k];
        s += v * v;
    }
    enorm[k] = s;
}

// ---------------------------------------------------------------------------
// K2: main VQ kernel. 256 blocks * 8 waves; block = 128 rows, wave = 16 rows.
// 64 code tiles of 16; score = 2*x.e - ||e||^2 via bf16x3 WMMA with three
// independent accumulator chains. B tiles double-buffered in LDS, staged by
// the Tensor Data Mover (wave 0 issues, s_wait_tensorcnt + barrier sync).
// ---------------------------------------------------------------------------
__global__ __launch_bounds__(256, 1)
void k_vq_argmin(const float* __restrict__ inputs,
                 const unsigned short* __restrict__ bHiG,
                 const unsigned short* __restrict__ bLoG,
                 const float* __restrict__ enorm,
                 int* __restrict__ idxOut) {
    __shared__ __align__(16) unsigned short BsHi[2][8 * 32 * 16];  // 2 x 8 KB
    __shared__ __align__(16) unsigned short BsLo[2][8 * 32 * 16];  // 2 x 8 KB
    __shared__ float enS[NUM_EMB];                                 // 4 KB

    const int tid   = threadIdx.x;
    const int lane  = tid & 31;
    const int wave  = tid >> 5;
    const int lhalf = lane >> 4;
    const int m     = lane & 15;   // A: row within wave tile; C: column n

    for (int i = tid; i < NUM_EMB; i += 256) enS[i] = enorm[i];

    const int n0 = blockIdx.x * 128;
    const int b  = n0 >> 10;
    const int hw = (n0 & 1023) + wave * 16 + m;
    const float* px = inputs + (size_t)b * (EMB_DIM * 1024) + hw;

    // A fragments (hi/lo bf16) resident in registers for the whole code loop.
    union FA { v16bf v; unsigned short u[16]; };
    FA aHi[8], aLo[8];
#pragma unroll
    for (int kc = 0; kc < 8; ++kc) {
#pragma unroll
        for (int s = 0; s < 16; ++s) {
            int cc = (s & 7) | ((s >> 3) << 4) | (lhalf << 3);  // ISA A layout
            float f = px[(kc * 32 + cc) * 1024];
            unsigned short h = f2bf(f);
            aHi[kc].u[s] = h;
            aLo[kc].u[s] = f2bf(f - bf2f(h));
        }
    }

    float best[8];
    int   bidx[8];
#pragma unroll
    for (int r = 0; r < 8; ++r) { best[r] = -3.4e38f; bidx[r] = 0; }

#if USE_TDM
    if (wave == 0) {   // prime the pipeline: tile 0 -> buffer 0
        tdm_load_8k(bHiG, (unsigned)(uintptr_t)&BsHi[0][0]);
        tdm_load_8k(bLoG, (unsigned)(uintptr_t)&BsLo[0][0]);
    }
#endif

    for (int t = 0; t < 64; ++t) {
        const int bb = t & 1;
#if USE_TDM
        if (wave == 0) __builtin_amdgcn_s_wait_tensorcnt(0);
        __syncthreads();                        // tile t resident in buffer bb
        if (wave == 0 && t + 1 < 64) {          // DMA next tile into other buf
            tdm_load_8k(bHiG + (t + 1) * 4096, (unsigned)(uintptr_t)&BsHi[bb ^ 1][0]);
            tdm_load_8k(bLoG + (t + 1) * 4096, (unsigned)(uintptr_t)&BsLo[bb ^ 1][0]);
        }
#else
        __syncthreads();
        {
            const uint4* sh = (const uint4*)(bHiG + t * 4096);
            const uint4* sl = (const uint4*)(bLoG + t * 4096);
            uint4* dh = (uint4*)&BsHi[bb][0];
            uint4* dl = (uint4*)&BsLo[bb][0];
            dh[tid * 2]     = sh[tid * 2];
            dh[tid * 2 + 1] = sh[tid * 2 + 1];
            dl[tid * 2]     = sl[tid * 2];
            dl[tid * 2 + 1] = sl[tid * 2 + 1];
        }
        __syncthreads();
#endif

        // three independent accumulator chains for WMMA ILP
        v8f acc0 = {0.f, 0.f, 0.f, 0.f, 0.f, 0.f, 0.f, 0.f};
        v8f acc1 = acc0, acc2 = acc0;
#pragma unroll
        for (int kc = 0; kc < 8; ++kc) {
            union FB { v16bf v; uint4 q[2]; } bh, bl;
            const uint4* ph = (const uint4*)&BsHi[bb][(kc * 32 + lane) * 16];
            const uint4* pl = (const uint4*)&BsLo[bb][(kc * 32 + lane) * 16];
            bh.q[0] = ph[0]; bh.q[1] = ph[1];
            bl.q[0] = pl[0]; bl.q[1] = pl[1];
            // x.e ~= xh*eh + xh*el + xl*eh   (f32 accumulate)
            acc0 = __builtin_amdgcn_wmma_f32_16x16x32_bf16(
                false, aHi[kc].v, false, bh.v, (short)0, acc0, false, false);
            acc1 = __builtin_amdgcn_wmma_f32_16x16x32_bf16(
                false, aHi[kc].v, false, bl.v, (short)0, acc1, false, false);
            acc2 = __builtin_amdgcn_wmma_f32_16x16x32_bf16(
                false, aLo[kc].v, false, bh.v, (short)0, acc2, false, false);
        }

        const float en   = enS[t * 16 + m];
        const int   code = t * 16 + m;
#pragma unroll
        for (int r = 0; r < 8; ++r) {
            float s2 = 2.0f * (acc0[r] + acc1[r] + acc2[r]) - en;
            if (s2 > best[r]) { best[r] = s2; bidx[r] = code; }
        }
    }

    // reduce over the 16 column-lanes of each half (ties -> smaller index,
    // matching argmin's first-hit semantics)
#pragma unroll
    for (int r = 0; r < 8; ++r) {
#pragma unroll
        for (int off = 8; off >= 1; off >>= 1) {
            float ob = __shfl_xor(best[r], off, 16);
            int   oi = __shfl_xor(bidx[r], off, 16);
            if (ob > best[r] || (ob == best[r] && oi < bidx[r])) {
                best[r] = ob; bidx[r] = oi;
            }
        }
    }
    if (m == 0) {
        int rowBase = n0 + wave * 16 + lhalf * 8;  // C layout: VGPR r -> M=r(+8)
#pragma unroll
        for (int r = 0; r < 8; ++r) idxOut[rowBase + r] = bidx[r];
    }
}

// ---------------------------------------------------------------------------
// K3: codebook gather -> out0 (NCHW) + per-block partial of sum((q-x)^2)
// ---------------------------------------------------------------------------
__global__ void k_quantize(const float* __restrict__ inputs,
                           const float* __restrict__ embed,
                           const int* __restrict__ idx,
                           float* __restrict__ out0,
                           float* __restrict__ partial) {
    __shared__ float red[256];
    int tid = threadIdx.x;
    int b   = blockIdx.x >> 2;
    int hw  = ((blockIdx.x & 3) << 8) + tid;
    int n   = b * 1024 + hw;
    int id  = idx[n];
    const float* xin = inputs + (size_t)b * (EMB_DIM * 1024) + hw;
    float*       xo  = out0   + (size_t)b * (EMB_DIM * 1024) + hw;
    float acc = 0.f;
    for (int c = 0; c < EMB_DIM; ++c) {
        float q = embed[c * NUM_EMB + id];   // embed stays L2-resident (1 MB)
        float x = xin[c * 1024];
        xo[c * 1024] = q;
        float d = q - x;
        acc += d * d;
    }
    red[tid] = acc;
    __syncthreads();
    for (int s = 128; s > 0; s >>= 1) {
        if (tid < s) red[tid] += red[tid + s];
        __syncthreads();
    }
    if (tid == 0) partial[blockIdx.x] = red[0];
}

// K4: deterministic final loss reduction (single block)
__global__ void k_loss(const float* __restrict__ partial,
                       float* __restrict__ loss) {
    __shared__ float red[128];
    int tid = threadIdx.x;
    red[tid] = partial[tid];
    __syncthreads();
    for (int s = 64; s > 0; s >>= 1) {
        if (tid < s) red[tid] += red[tid + s];
        __syncthreads();
    }
    if (tid == 0) *loss = 1.25f * red[0] / (float)OUT0_ELEMS;
}

// ---------------------------------------------------------------------------
// K5: one-hot writer. One wave per row; each lane owns 32 columns, written as
// 8 float4 stores (this 128 MB stream is the bandwidth floor of the problem).
// ---------------------------------------------------------------------------
__global__ void k_onehot(const int* __restrict__ idx,
                         float* __restrict__ onehot) {
    int lane = threadIdx.x & 31;
    int wave = threadIdx.x >> 5;
    int row  = blockIdx.x * 8 + wave;
    int id   = idx[row];
    float4* p = (float4*)(onehot + (size_t)row * NUM_EMB + lane * 32);
#pragma unroll
    for (int v = 0; v < 8; ++v) {
        int pos = lane * 32 + v * 4;
        float4 z = {0.f, 0.f, 0.f, 0.f};
        int d = id - pos;
        if (d >= 0 && d < 4) ((float*)&z)[d] = 1.0f;
        p[v] = z;
    }
}

extern "C" void kernel_launch(void* const* d_in, const int* in_sizes, int n_in,
                              void* d_out, int out_size, void* d_ws, size_t ws_size,
                              hipStream_t stream) {
    const float* inputs = (const float*)d_in[0];   // (32,256,32,32) f32
    const float* embed  = (const float*)d_in[1];   // (256,1024)     f32

    float* out0   = (float*)d_out;                 // quantize_st (NCHW)
    float* loss   = out0 + OUT0_ELEMS;             // scalar
    float* onehot = loss + 1;                      // (32768,1024)

    char* ws = (char*)d_ws;
    unsigned short* bHi   = (unsigned short*)(ws);             // 512 KB
    unsigned short* bLo   = (unsigned short*)(ws + 524288);    // 512 KB
    float*          enorm = (float*)(ws + 1048576);            //   4 KB
    int*            idx   = (int*)(ws + 1052672);              // 128 KB
    float*          part  = (float*)(ws + 1183744);            // 512 B

    k_embed_swizzle<<<1024, 256, 0, stream>>>(embed, bHi, bLo);
    k_embed_norm<<<4, 256, 0, stream>>>(embed, enorm);
    k_vq_argmin<<<256, 256, 0, stream>>>(inputs, bHi, bLo, enorm, idx);
    k_quantize<<<128, 256, 0, stream>>>(inputs, embed, idx, out0, part);
    k_loss<<<1, 128, 0, stream>>>(part, loss);
    k_onehot<<<4096, 256, 0, stream>>>(idx, onehot);
}
